// HyperbolicGraphConvolution_59124519796865
// MI455X (gfx1250) — matrix operand order repaired and checked by reading, not compiled, not run on hardware
//
#include <hip/hip_runtime.h>

#define EPSF 1e-7f

typedef __bf16 bf16_t;
typedef __attribute__((ext_vector_type(16))) __bf16 v16bf;
typedef __attribute__((ext_vector_type(8)))  float  v8f;

// ---------- scalar math helpers ----------
__device__ __forceinline__ float artanh_c(float x) {
    x = fminf(fmaxf(x, -1.0f + EPSF), 1.0f - EPSF);
    return 0.5f * __logf((1.0f + x) / (1.0f - x));
}
__device__ __forceinline__ float sigmoidf_(float x) { return 1.0f / (1.0f + __expf(-x)); }
__device__ __forceinline__ float siluf_(float x) { return x * sigmoidf_(x); }

// butterfly sum across the 16 lanes of a half-wave (masks <= 8 stay in half)
__device__ __forceinline__ float red16(float x) {
    x += __shfl_xor(x, 8, 32);
    x += __shfl_xor(x, 4, 32);
    x += __shfl_xor(x, 2, 32);
    x += __shfl_xor(x, 1, 32);
    return x;
}

// ---------- WMMA fragment builders (wave32, 16x16x32 bf16) ----------
// A (16x32, MxK): lane L holds row M=L%16; elems 0..7 = K(koff + (L/16)*16 + j),
// elems 8..15 = K(koff + (L/16)*16 + 8 + j)
__device__ __forceinline__ v16bf load_a_bf16(const bf16_t* rowp, int koff, int hi8) {
    v16bf a;
#pragma unroll
    for (int j = 0; j < 8; ++j) {
        a[j]     = rowp[koff + 2 * hi8 + j];
        a[8 + j] = rowp[koff + 2 * hi8 + 8 + j];
    }
    return a;
}
__device__ __forceinline__ v16bf load_a_f32(const float* rowp, int koff, int hi8) {
    v16bf a;
#pragma unroll
    for (int j = 0; j < 8; ++j) {
        a[j]     = (bf16_t)rowp[koff + 2 * hi8 + j];
        a[8 + j] = (bf16_t)rowp[koff + 2 * hi8 + 8 + j];
    }
    return a;
}
// B (32x16, KxN) from row-major W[nout][ktot]: B[k][n] = W[n][k].
// Lane L holds col N=L%16, elems j = K(kglob_base + j), kglob_base includes (L/16)*16.
__device__ __forceinline__ v16bf load_b_f32(const float* W, int ktot, int nglob, int kglob_base) {
    v16bf b;
    const float* p = W + (size_t)nglob * ktot + kglob_base;
#pragma unroll
    for (int j = 0; j < 16; ++j) b[j] = (bf16_t)p[j];
    return b;
}
// B fragment from pre-swizzled bf16 buffer: [frag][lane][16] contiguous
__device__ __forceinline__ v16bf load_b_pre(const bf16_t* Wf, int frag, int lane) {
    v16bf b;
    const bf16_t* p = Wf + ((size_t)frag * 32 + lane) * 16;
#pragma unroll
    for (int j = 0; j < 16; ++j) b[j] = p[j];
    return b;
}
__device__ __forceinline__ v8f wmma_bf16(v16bf a, v16bf b, v8f c) {
    return __builtin_amdgcn_wmma_f32_16x16x32_bf16(false, a, false, b, (short)0, c, false, false);
}

// ======================================================================
// Kernel 0: pre-swizzle Wa1 (first 128 K) into per-lane bf16 B-fragment
// layout: frag f = nt*4+ch ; lane L -> col N=L%16, kbase = ch*32+(L/16)*16
// ======================================================================
__global__ void __launch_bounds__(512) k_prep_wa1(
    const float* __restrict__ Wa1, bf16_t* __restrict__ Wa1f)
{
    int t = threadIdx.x;            // 0..511 = frag*32 + lane
    int frag = t >> 5, lane = t & 31;
    int nt = frag >> 2, ch = frag & 3;
    int ng = nt * 16 + (lane & 15);
    int kb = ch * 32 + (lane >> 4) * 16;
    const float* p = Wa1 + (size_t)ng * 130 + kb;
    bf16_t* q = Wa1f + (size_t)t * 16;
#pragma unroll
    for (int j = 0; j < 16; ++j) q[j] = (bf16_t)p[j];
}

// ======================================================================
// Kernel 1: node pre-pass.  xt = logmap0(h) @ W_lin.T ; res = expmap0(xt);
// hb = expmap(res, transp0(res, b_lin)); store hb (bf16) and x_tan = logmap0(hb)
// ======================================================================
__global__ void __launch_bounds__(256) k_node_pre(
    const float* __restrict__ h, const float* __restrict__ W_lin,
    const float* __restrict__ b_lin,
    float* __restrict__ x_tan, bf16_t* __restrict__ hb16, int nnodes)
{
    int wave = (blockIdx.x * blockDim.x + threadIdx.x) >> 5;
    int lane = threadIdx.x & 31;
    int ntiles = nnodes >> 4;
    if (wave >= ntiles) return;
    int m = lane & 15, hi = lane >> 4, hi8 = hi * 8;
    const float* hr = h + (size_t)(wave * 16 + m) * 64;

    // load this lane's 32 K-subset elements of u, accumulate half-norm
    float uf[32];
    float n2p = 0.f;
#pragma unroll
    for (int c2 = 0; c2 < 2; ++c2) {
#pragma unroll
        for (int j = 0; j < 8; ++j) {
            float a = hr[c2 * 32 + 2 * hi8 + j];
            float b = hr[c2 * 32 + 2 * hi8 + 8 + j];
            uf[c2 * 16 + j] = a; uf[c2 * 16 + 8 + j] = b;
            n2p += a * a + b * b;
        }
    }
    float n2 = n2p + __shfl_xor(n2p, 16, 32);
    float nn = fmaxf(sqrtf(n2), EPSF);
    float sc = artanh_c(nn) / nn;          // logmap0 scale

    v16bf a0, a1;
#pragma unroll
    for (int j = 0; j < 16; ++j) { a0[j] = (bf16_t)(uf[j] * sc); a1[j] = (bf16_t)(uf[16 + j] * sc); }

    v8f C[4];
#pragma unroll
    for (int nt = 0; nt < 4; ++nt) {
        int ng = nt * 16 + m;
        v8f acc = {};
        acc = wmma_bf16(a0, load_b_f32(W_lin, 64, ng, 0  + hi * 16), acc);
        acc = wmma_bf16(a1, load_b_f32(W_lin, 64, ng, 32 + hi * 16), acc);
        C[nt] = acc;
    }

    // expmap0 on each row (C layout: lane holds M = v + 8*hi, N = nt*16 + m)
    float res2[8];
#pragma unroll
    for (int v = 0; v < 8; ++v) {
        float p = 0.f;
#pragma unroll
        for (int nt = 0; nt < 4; ++nt) p += C[nt][v] * C[nt][v];
        float s2 = red16(p);
        float n = fmaxf(sqrtf(s2), EPSF);
        float s = tanhf(n) / n;
#pragma unroll
        for (int nt = 0; nt < 4; ++nt) C[nt][v] *= s;
        res2[v] = s * s * s2;
    }

    // hyperbolic bias: hb = mobius_add(res, tanh(0.5*lam*nu) * u/nu), u = b*(1-res2)
    float bn2 = 0.f;
    for (int k = 0; k < 64; ++k) { float b = b_lin[k]; bn2 += b * b; }
    float bnorm = sqrtf(bn2);
    float bb[4];
#pragma unroll
    for (int nt = 0; nt < 4; ++nt) bb[nt] = b_lin[nt * 16 + m];

#pragma unroll
    for (int v = 0; v < 8; ++v) {
        float fac = 1.0f - res2[v];
        float nu  = fmaxf(fabsf(fac) * bnorm, EPSF);
        float lam = 2.0f / fmaxf(fac, EPSF);
        float t   = tanhf(0.5f * lam * nu);
        float ys  = t * fac / nu;          // y = ys * b_lin
        float y2  = ys * ys * bn2;
        float p = 0.f;
#pragma unroll
        for (int nt = 0; nt < 4; ++nt) p += C[nt][v] * bb[nt];
        float xy = ys * red16(p);
        float A1 = 1.0f + 2.0f * xy + y2;
        float B1 = 1.0f - res2[v];
        float den = fmaxf(1.0f + 2.0f * xy + res2[v] * y2, EPSF);
        float ia = A1 / den, ib = B1 * ys / den;
#pragma unroll
        for (int nt = 0; nt < 4; ++nt) C[nt][v] = ia * C[nt][v] + ib * bb[nt];
    }

    // x_tan = logmap0(hb); store hb (bf16) and x_tan (f32)
#pragma unroll
    for (int v = 0; v < 8; ++v) {
        float p = 0.f;
#pragma unroll
        for (int nt = 0; nt < 4; ++nt) p += C[nt][v] * C[nt][v];
        float h2 = red16(p);
        float n = fmaxf(sqrtf(h2), EPSF);
        float s = artanh_c(n) / n;
        size_t row = (size_t)(wave * 16 + v + hi8);
#pragma unroll
        for (int nt = 0; nt < 4; ++nt) {
            int colg = nt * 16 + m;
            float hv = C[nt][v];
            hb16[row * 64 + colg]  = (bf16_t)hv;
            x_tan[row * 64 + colg] = s * hv;
        }
    }
}

// ======================================================================
// Kernel 2: edge attention + scatter. One wave = 16 edges; TPW tiles/wave to
// amortize the Wa1 B-fragments held in registers (loaded pre-swizzled).
// ======================================================================
#define TPW 4
__global__ void __launch_bounds__(256) k_edge(
    const bf16_t* __restrict__ hb16, const float* __restrict__ x_tan,
    const int* __restrict__ edges, const float* __restrict__ distances,
    const float* __restrict__ edge_mask,
    const bf16_t* __restrict__ Wa1f, const float* __restrict__ Wa1,
    const float* __restrict__ ba1,
    const float* __restrict__ Wa2, const float* __restrict__ ba2,
    float* __restrict__ agg, int nedges)
{
    int wave = (blockIdx.x * blockDim.x + threadIdx.x) >> 5;
    int lane = threadIdx.x & 31;
    int etiles = nedges >> 4;
    int t0 = wave * TPW;
    if (t0 >= etiles) return;
    int m = lane & 15, hi = lane >> 4, hi8 = hi * 8;

    // all B-fragments of Wa1 (4 n-tiles x 4 K-chunks), pre-swizzled bf16
    v16bf Bf[16];
#pragma unroll
    for (int f = 0; f < 16; ++f) Bf[f] = load_b_pre(Wa1f, f, lane);
    float wd[4], wdist[4], bias1[4], w2[4];
#pragma unroll
    for (int nt = 0; nt < 4; ++nt) {
        int ng = nt * 16 + m;
        wd[nt]    = Wa1[(size_t)ng * 130 + 128];
        wdist[nt] = Wa1[(size_t)ng * 130 + 129];
        bias1[nt] = ba1[ng];
        w2[nt]    = Wa2[ng];
    }
    float b2 = ba2[0];

    for (int t = 0; t < TPW; ++t) {
        int tile = t0 + t;
        if (tile >= etiles) break;
        int e = tile * 16 + m;
        int r = edges[e], c = edges[nedges + e];

        const bf16_t* xr = hb16 + (size_t)r * 64;
        const bf16_t* xc = hb16 + (size_t)c * 64;
        v16bf ar0 = load_a_bf16(xr, 0, hi8), ar1 = load_a_bf16(xr, 32, hi8);
        v16bf ac0 = load_a_bf16(xc, 0, hi8), ac1 = load_a_bf16(xc, 32, hi8);

        // dots for pdist, straight from the fragments (each lane holds half of K)
        float x2p = 0.f, y2p = 0.f, xyp = 0.f;
#pragma unroll
        for (int j = 0; j < 16; ++j) {
            float xa = (float)ar0[j], xb = (float)ar1[j];
            float ya = (float)ac0[j], yb = (float)ac1[j];
            x2p += xa * xa + xb * xb;
            y2p += ya * ya + yb * yb;
            xyp += xa * ya + xb * yb;
        }
        float x2 = x2p + __shfl_xor(x2p, 16, 32);
        float y2 = y2p + __shfl_xor(y2p, 16, 32);
        float xy = xyp + __shfl_xor(xyp, 16, 32);
        // mobius_add(-x, y): num = -(1-2xy+y2)x + (1-x2)y ; den = 1-2xy+x2*y2
        float A1 = 1.0f - 2.0f * xy + y2;
        float B1 = 1.0f - x2;
        float den = fmaxf(1.0f - 2.0f * xy + x2 * y2, EPSF);
        float num2 = fmaxf(A1 * A1 * x2 - 2.0f * A1 * B1 * xy + B1 * B1 * y2, 0.0f);
        float dloc = 2.0f * artanh_c(sqrtf(num2) / den);
        float distloc = distances[e];
        float emloc = edge_mask[e];

        // broadcast per-row edge features into C layout once (M = v + 8*hi)
        float dMv[8], diMv[8];
#pragma unroll
        for (int v = 0; v < 8; ++v) {
            dMv[v]  = __shfl(dloc,    v + hi8, 32);
            diMv[v] = __shfl(distloc, v + hi8, 32);
        }

        float spart[8] = {};
#pragma unroll
        for (int nt = 0; nt < 4; ++nt) {
            v8f acc = {};
            acc = wmma_bf16(ar0, Bf[nt * 4 + 0], acc);
            acc = wmma_bf16(ar1, Bf[nt * 4 + 1], acc);
            acc = wmma_bf16(ac0, Bf[nt * 4 + 2], acc);
            acc = wmma_bf16(ac1, Bf[nt * 4 + 3], acc);
#pragma unroll
            for (int v = 0; v < 8; ++v) {
                float att = acc[v] + dMv[v] * wd[nt] + diMv[v] * wdist[nt] + bias1[nt];
                spart[v] += siluf_(att) * w2[nt];
            }
        }
        float score[8];
#pragma unroll
        for (int v = 0; v < 8; ++v) {
            float s  = red16(spart[v]);
            float em = __shfl(emloc, v + hi8, 32);
            score[v] = sigmoidf_(s + b2) * em * 0.01f;   // fold /NORM_FACTOR
        }

        // scatter: agg[row] += x_tan[col] * score  (whole wave per edge, 2 cols/lane)
#pragma unroll
        for (int el = 0; el < 16; ++el) {
            float scE = __shfl(score[el & 7], (el >> 3) * 16, 32);
            int rE = __shfl(r, el, 32);
            int cE = __shfl(c, el, 32);
            const float* src = x_tan + (size_t)cE * 64;
            float* dst = agg + (size_t)rE * 64;
            atomicAdd(&dst[lane],      src[lane]      * scE);
            atomicAdd(&dst[lane + 32], src[lane + 32] * scE);
        }
    }
}

// ======================================================================
// Kernel 3: node post-pass. mlp = silu(agg@Wm1.T+bm1)@Wm2.T+bm2 ;
// out = expmap0(x_tan + mlp) ; final = expmap0(silu(logmap0(out)))
// Single-wave blocks: LDS re-fragmentation, barrier is a NOP.
// ======================================================================
__global__ void __launch_bounds__(32) k_node_post(
    const float* __restrict__ agg, const float* __restrict__ x_tan,
    const float* __restrict__ Wm1, const float* __restrict__ bm1,
    const float* __restrict__ Wm2, const float* __restrict__ bm2,
    float* __restrict__ out, int nnodes)
{
    __shared__ bf16_t lds[16 * 64];
    int wave = blockIdx.x;
    int lane = threadIdx.x & 31;
    int ntiles = nnodes >> 4;
    if (wave >= ntiles) return;
    int m = lane & 15, hi = lane >> 4, hi8 = hi * 8;
    const float* ar = agg + (size_t)(wave * 16 + m) * 64;

    v16bf a0 = load_a_f32(ar, 0, hi8);
    v16bf a1 = load_a_f32(ar, 32, hi8);

    // hidden = silu(agg@Wm1.T + bm1) -> LDS tile (row-major 16x64 bf16)
#pragma unroll
    for (int nt = 0; nt < 4; ++nt) {
        int ng = nt * 16 + m;
        v8f acc = {};
        acc = wmma_bf16(a0, load_b_f32(Wm1, 64, ng, 0  + hi * 16), acc);
        acc = wmma_bf16(a1, load_b_f32(Wm1, 64, ng, 32 + hi * 16), acc);
        float bias = bm1[ng];
#pragma unroll
        for (int v = 0; v < 8; ++v)
            lds[(v + hi8) * 64 + ng] = (bf16_t)siluf_(acc[v] + bias);
    }
    __syncthreads();
    v16bf h0 = load_a_bf16(&lds[m * 64], 0, hi8);
    v16bf h1 = load_a_bf16(&lds[m * 64], 32, hi8);

    v8f C2[4];
#pragma unroll
    for (int nt = 0; nt < 4; ++nt) {
        int ng = nt * 16 + m;
        v8f acc = {};
        acc = wmma_bf16(h0, load_b_f32(Wm2, 64, ng, 0  + hi * 16), acc);
        acc = wmma_bf16(h1, load_b_f32(Wm2, 64, ng, 32 + hi * 16), acc);
        float bias = bm2[ng];
#pragma unroll
        for (int v = 0; v < 8; ++v)
            acc[v] += bias + x_tan[(size_t)(wave * 16 + v + hi8) * 64 + ng];
        C2[nt] = acc;
    }

    // out = expmap0(z); act = silu(logmap0(out)); final = expmap0(act)
#pragma unroll
    for (int v = 0; v < 8; ++v) {
        float p = 0.f;
#pragma unroll
        for (int nt = 0; nt < 4; ++nt) p += C2[nt][v] * C2[nt][v];
        float z2 = red16(p);
        float nz = fmaxf(sqrtf(z2), EPSF);
        float s1 = tanhf(nz) / nz;                         // expmap0 scale
        float no = fmaxf(s1 * nz, EPSF);                   // ||out|| = tanh(nz)
        float s2 = artanh_c(no) / no * s1;                 // logmap0(out) scale wrt z
        float l[4];
        float ap = 0.f;
#pragma unroll
        for (int nt = 0; nt < 4; ++nt) { l[nt] = siluf_(s2 * C2[nt][v]); ap += l[nt] * l[nt]; }
        float a2 = red16(ap);
        float na = fmaxf(sqrtf(a2), EPSF);
        float s3 = tanhf(na) / na;
        size_t row = (size_t)(wave * 16 + v + hi8);
#pragma unroll
        for (int nt = 0; nt < 4; ++nt)
            out[row * 64 + nt * 16 + m] = s3 * l[nt];
    }
}

// ======================================================================
extern "C" void kernel_launch(void* const* d_in, const int* in_sizes, int n_in,
                              void* d_out, int out_size, void* d_ws, size_t ws_size,
                              hipStream_t stream)
{
    const float* h     = (const float*)d_in[0];
    const float* dists = (const float*)d_in[1];
    const int*   edges = (const int*)d_in[2];
    // d_in[3] node_mask: unused by the reference
    const float* emask = (const float*)d_in[4];
    const float* W_lin = (const float*)d_in[5];
    const float* b_lin = (const float*)d_in[6];
    const float* Wa1   = (const float*)d_in[7];
    const float* ba1   = (const float*)d_in[8];
    const float* Wa2   = (const float*)d_in[9];
    const float* ba2   = (const float*)d_in[10];
    const float* Wm1   = (const float*)d_in[11];
    const float* bm1   = (const float*)d_in[12];
    const float* Wm2   = (const float*)d_in[13];
    const float* bm2   = (const float*)d_in[14];

    const int nnodes = in_sizes[0] / 64;   // 50000
    const int nedges = in_sizes[1];        // 800000

    char* ws = (char*)d_ws;
    float*  agg   = (float*)ws;                                      // N*64 f32
    float*  xtan  = (float*)(ws + (size_t)nnodes * 64 * 4);          // N*64 f32
    bf16_t* hb16  = (bf16_t*)(ws + (size_t)nnodes * 64 * 8);         // N*64 bf16
    bf16_t* Wa1f  = (bf16_t*)(ws + (size_t)nnodes * 64 * 10);        // 16KB fragments

    hipMemsetAsync(agg, 0, (size_t)nnodes * 64 * sizeof(float), stream);

    int ntiles = nnodes / 16;
    int etiles = nedges / 16;
    int ewaves = (etiles + TPW - 1) / TPW;

    k_prep_wa1<<<1, 512, 0, stream>>>(Wa1, Wa1f);
    k_node_pre<<<(ntiles + 7) / 8, 256, 0, stream>>>(h, W_lin, b_lin, xtan, hb16, nnodes);
    k_edge<<<(ewaves + 7) / 8, 256, 0, stream>>>(hb16, xtan, edges, dists, emask,
                                                 Wa1f, Wa1, ba1, Wa2, ba2, agg, nedges);
    k_node_post<<<ntiles, 32, 0, stream>>>(agg, xtan, Wm1, bm1, Wm2, bm2,
                                           (float*)d_out, nnodes);
}
